// GraphEncoder_31301721653959
// MI455X (gfx1250) — compile-verified, matrix operand order checked by reading, not run on hardware
//
#include <hip/hip_runtime.h>
#include <cstring>
#include <cstdint>

// ---------------------------------------------------------------------------
// Types for CDNA5 WMMA (wave32): v_wmma_f32_16x16x32_bf16
// ---------------------------------------------------------------------------
typedef __attribute__((ext_vector_type(16))) __bf16 v16bf;
typedef __attribute__((ext_vector_type(8)))  __bf16 v8bf;
typedef __attribute__((ext_vector_type(8)))  float  v8f;

// Orderable-uint encoding of float for atomicMax-based segment max
static __device__ __forceinline__ unsigned enc_f32(float x) {
  unsigned u = __float_as_uint(x);
  return u ^ ((u >> 31) ? 0xFFFFFFFFu : 0x80000000u);
}
static __device__ __forceinline__ float dec_f32(unsigned k) {
  unsigned u = (k >> 31) ? (k ^ 0x80000000u) : ~k;
  return __uint_as_float(u);
}
static __device__ __forceinline__ unsigned short f32_to_bf16_rne(float f) {
  unsigned u = __float_as_uint(f);
  u += 0x7FFFu + ((u >> 16) & 1u);
  return (unsigned short)(u >> 16);
}

// ---------------------------------------------------------------------------
// Utility kernels
// ---------------------------------------------------------------------------
__global__ void k_fill_f32(float* p, float v, size_t n) {
  size_t i = (size_t)blockIdx.x * blockDim.x + threadIdx.x;
  if (i < n) p[i] = v;
}
__global__ void k_fill_u32(unsigned* p, unsigned v, size_t n) {
  size_t i = (size_t)blockIdx.x * blockDim.x + threadIdx.x;
  if (i < n) p[i] = v;
}
__global__ void k_cvt_bf16(const float* __restrict__ src, unsigned short* __restrict__ dst, size_t n) {
  size_t i = (size_t)blockIdx.x * blockDim.x + threadIdx.x;
  if (i < n) dst[i] = f32_to_bf16_rne(src[i]);
}
__global__ void k_relu_cvt_bf16(const float* __restrict__ src, unsigned short* __restrict__ dst, size_t n) {
  size_t i = (size_t)blockIdx.x * blockDim.x + threadIdx.x;
  if (i < n) {
    float v = src[i];
    dst[i] = f32_to_bf16_rne(v > 0.0f ? v : 0.0f);
  }
}

// ---------------------------------------------------------------------------
// Pack a row-major f32 weight matrix B[K,Nc] into WMMA B-fragment layout
// (bf16), so each wave can fetch a whole 16-half fragment with one 32-byte
// contiguous load.  Layout:
//   Bpack[((ntile*nk + kk)*32 + lane)*16 + i]
// where per CDNA5 ISA 7.12.2 (B 32x16, 16-bit):
//   col = ntile*16 + (lane&15), kh = lane>>4,
//   i in 0..7  -> K = kk*32 + 8*kh + i
//   i in 8..15 -> K = kk*32 + 16 + 8*kh + (i-8)
// ---------------------------------------------------------------------------
__global__ void k_packB(const float* __restrict__ B, unsigned short* __restrict__ Bpack,
                        int K, int Nc)
{
  int idx = blockIdx.x * blockDim.x + threadIdx.x;       // one thread per fragment-lane
  const int nk = K >> 5;
  const int total = (Nc >> 4) * nk * 32;
  if (idx >= total) return;
  const int lane = idx & 31;
  const int kk   = (idx >> 5) % nk;
  const int t    = (idx >> 5) / nk;
  const int col  = t * 16 + (lane & 15);
  const int kh   = lane >> 4;
  unsigned short* dst = Bpack + (size_t)idx * 16;
  const int kbase = kk * 32 + 8 * kh;
  #pragma unroll
  for (int i = 0; i < 8; ++i)
    dst[i]     = f32_to_bf16_rne(B[(size_t)(kbase + i) * Nc + col]);
  #pragma unroll
  for (int i = 0; i < 8; ++i)
    dst[8 + i] = f32_to_bf16_rne(B[(size_t)(kbase + 16 + i) * Nc + col]);
}

// ---------------------------------------------------------------------------
// WMMA GEMM: Out[M,Nc] = A[M,K](bf16, row-major) @ B(packed) (+bias[Nc])
// Compile-time K -> fully unrolled K loop; NT=4 column tiles per wave
// (16x64 output per wave) so the A fragment is reused across 4 WMMAs.
// ---------------------------------------------------------------------------
template <int K, int NT>
__global__ void k_gemm_bf16(const unsigned short* __restrict__ A,
                            const unsigned short* __restrict__ Bpack,
                            const float* __restrict__ bias,
                            float* __restrict__ Out,
                            int M, int Nc)
{
  constexpr int NK = K >> 5;
  const int wave  = threadIdx.x >> 5;
  const int lane  = threadIdx.x & 31;
  const int mtile = blockIdx.x * (blockDim.x >> 5) + wave;
  const int row0  = mtile * 16;
  const int nt0   = blockIdx.y * NT;        // first column tile
  const int col0  = nt0 * 16;
  if (row0 >= M) return;                    // wave-uniform exit (EXEC stays full)

  const int col  = lane & 15;
  const int kh   = lane >> 4;               // 0 or 1
  const int arow = row0 + col;              // A-row for this lane (M = lane%16)
  const bool aok = (arow < M);

  const unsigned short* arowp = A + (size_t)arow * K;
  if (aok) __builtin_prefetch(arowp, 0, 0); // global_prefetch_b8

  v8f acc[NT];
  #pragma unroll
  for (int t = 0; t < NT; ++t) acc[t] = (v8f){};

  #pragma unroll
  for (int kk = 0; kk < NK; ++kk) {
    v16bf a;
    if (aok) {
      const unsigned short* ap = arowp + kk * 32 + kh * 8;
      v8bf lo = *reinterpret_cast<const v8bf*>(ap);        // K = 8kh .. 8kh+7
      v8bf hi = *reinterpret_cast<const v8bf*>(ap + 16);   // K = 16+8kh .. +7
      a = __builtin_shufflevector(lo, hi, 0, 1, 2, 3, 4, 5, 6, 7,
                                  8, 9, 10, 11, 12, 13, 14, 15);
    } else {
      a = (v16bf){};
    }
    #pragma unroll
    for (int t = 0; t < NT; ++t) {
      const v16bf b = *reinterpret_cast<const v16bf*>(
          Bpack + ((size_t)((nt0 + t) * NK + kk) * 32 + lane) * 16);
      acc[t] = __builtin_amdgcn_wmma_f32_16x16x32_bf16(false, a, false, b,
                                                       (short)0, acc[t], false, false);
    }
  }

  #pragma unroll
  for (int t = 0; t < NT; ++t) {
    const int c = col0 + t * 16 + col;
    const float bv = bias ? bias[c] : 0.0f;
    #pragma unroll
    for (int v = 0; v < 8; ++v) {
      int r = row0 + v + kh * 8;
      if (r < M) Out[(size_t)r * Nc + c] = acc[t][v] + bv;
    }
  }
}

// ---------------------------------------------------------------------------
// Relation histogram + mean edge-feature projection
// ---------------------------------------------------------------------------
__global__ void k_counts(const int* __restrict__ ridx, float* __restrict__ counts, int E) {
  int e = blockIdx.x * blockDim.x + threadIdx.x;
  if (e < E) atomicAdd(&counts[ridx[e]], 1.0f);
}
__global__ void k_meanp(const float* __restrict__ counts, const float* __restrict__ relp,
                        float* __restrict__ meanp, int R, float invE) {
  int j = threadIdx.x;                    // 256 columns
  float s = 0.0f;
  for (int r = 0; r < R; ++r) s += counts[r] * relp[(size_t)r * 256 + j];
  meanp[j] = s * invE;
}

// ---------------------------------------------------------------------------
// Per-edge attention logits (warp per edge) + segment max via atomicMax(key)
// ---------------------------------------------------------------------------
__global__ void k_alpha(const int* __restrict__ ei, const int* __restrict__ ridx,
                        const float* __restrict__ xl, const float* __restrict__ xr,
                        const float* __restrict__ relp, const float* __restrict__ meanp,
                        const float* __restrict__ att,
                        float* __restrict__ alpha, unsigned* __restrict__ amaxk,
                        int E, int EN)
{
  int w    = (int)(((size_t)blockIdx.x * blockDim.x + threadIdx.x) >> 5);
  int lane = threadIdx.x & 31;
  if (w >= EN) return;

  int s, d; const float* ea;
  if (w < E) { s = ei[w]; d = ei[E + w]; ea = relp + (size_t)ridx[w] * 256; }
  else       { s = d = w - E;            ea = meanp; }

  const float* ps = xl + (size_t)s * 256;
  const float* pd = xr + (size_t)d * 256;
  float a0 = 0.0f, a1 = 0.0f;
  for (int j = lane; j < 256; j += 32) {
    float m = ps[j] + pd[j] + ea[j];
    m = (m > 0.0f) ? m : 0.2f * m;        // leaky_relu, NEG_SLOPE = 0.2
    float v = att[j] * m;
    if (j < 128) a0 += v; else a1 += v;
  }
  for (int o = 16; o > 0; o >>= 1) {
    a0 += __shfl_xor(a0, o, 32);
    a1 += __shfl_xor(a1, o, 32);
  }
  if (lane == 0) {
    alpha[2 * (size_t)w]     = a0;
    alpha[2 * (size_t)w + 1] = a1;
    atomicMax(&amaxk[2 * (size_t)d],     enc_f32(a0));
    atomicMax(&amaxk[2 * (size_t)d + 1], enc_f32(a1));
  }
}

__global__ void k_expsum(const int* __restrict__ ei, float* __restrict__ alpha,
                         const unsigned* __restrict__ amaxk, float* __restrict__ denom,
                         int E, int EN)
{
  int e = blockIdx.x * blockDim.x + threadIdx.x;
  if (e >= EN) return;
  int d = (e < E) ? ei[E + e] : (e - E);
  #pragma unroll
  for (int h = 0; h < 2; ++h) {
    float v = __expf(alpha[2 * (size_t)e + h] - dec_f32(amaxk[2 * (size_t)d + h]));
    alpha[2 * (size_t)e + h] = v;
    atomicAdd(&denom[2 * (size_t)d + h], v);
  }
}

// out[d,c] += 0.5*(alpha0/denom0 * xl[s,c] + alpha1/denom1 * xl[s,128+c])
__global__ void k_scatter(const int* __restrict__ ei, const float* __restrict__ xl,
                          const float* __restrict__ alpha, const float* __restrict__ denom,
                          float* __restrict__ out, int E, int EN)
{
  int w    = (int)(((size_t)blockIdx.x * blockDim.x + threadIdx.x) >> 5);
  int lane = threadIdx.x & 31;
  if (w >= EN) return;
  int s, d;
  if (w < E) { s = ei[w]; d = ei[E + w]; } else { s = d = w - E; }
  float w0 = alpha[2 * (size_t)w]     / denom[2 * (size_t)d];
  float w1 = alpha[2 * (size_t)w + 1] / denom[2 * (size_t)d + 1];
  const float* xs = xl  + (size_t)s * 256;
  float*       od = out + (size_t)d * 128;
  for (int c = lane; c < 128; c += 32) {
    float v = 0.5f * (w0 * xs[c] + w1 * xs[128 + c]);
    atomicAdd(&od[c], v);
  }
}

__global__ void k_biasadd(float* __restrict__ out, const float* __restrict__ bias, size_t n, int C) {
  size_t i = (size_t)blockIdx.x * blockDim.x + threadIdx.x;
  if (i < n) out[i] += bias[i & (size_t)(C - 1)];
}

// ---------------------------------------------------------------------------
// Host-side launch
// ---------------------------------------------------------------------------
extern "C" void kernel_launch(void* const* d_in, const int* in_sizes, int n_in,
                              void* d_out, int out_size, void* d_ws, size_t ws_size,
                              hipStream_t stream)
{
  (void)in_sizes; (void)n_in; (void)out_size; (void)ws_size;
  const int N = 50000, E = 400000, D = 128, HC = 256, R = 500, C = 128;
  const int EN = E + N;

  const float* x      = (const float*)d_in[0];
  const int*   ei     = (const int*)  d_in[1];   // [2,E] flat: src=ei[e], dst=ei[E+e]
  const float* rels   = (const float*)d_in[2];
  const int*   ridx   = (const int*)  d_in[3];
  const float* W_l    = (const float*)d_in[4];
  const float* b_l    = (const float*)d_in[5];
  const float* W_r    = (const float*)d_in[6];
  const float* b_r    = (const float*)d_in[7];
  const float* W_e    = (const float*)d_in[8];
  const float* att    = (const float*)d_in[9];
  const float* bias_g = (const float*)d_in[10];
  const float* W_le   = (const float*)d_in[11];
  const float* b_le   = (const float*)d_in[12];

  float* out_nodes = (float*)d_out;                    // [N, C]
  float* out_rel   = (float*)d_out + (size_t)N * C;    // [R, C]

  // ---- carve workspace ----
  char* ws = (char*)d_ws;
  size_t off = 0;
  auto carve = [&](size_t bytes) -> char* {
    char* p = ws + off;
    off += (bytes + 255) & ~(size_t)255;
    return p;
  };
  float*          xl      = (float*)          carve((size_t)N * HC * 4);
  float*          xr      = (float*)          carve((size_t)N * HC * 4);
  float*          relp    = (float*)          carve((size_t)512 * HC * 4);
  float*          meanp   = (float*)          carve((size_t)HC * 4);
  float*          counts  = (float*)          carve((size_t)R * 4);
  float*          alpha   = (float*)          carve((size_t)EN * 2 * 4);
  unsigned*       amaxk   = (unsigned*)       carve((size_t)N * 2 * 4);
  float*          denom   = (float*)          carve((size_t)N * 2 * 4);
  unsigned short* xbf     = (unsigned short*) carve((size_t)N * D * 2);
  unsigned short* relbf   = (unsigned short*) carve((size_t)R * D * 2);
  unsigned short* WlPk    = (unsigned short*) carve((size_t)D * HC * 2);
  unsigned short* WrPk    = (unsigned short*) carve((size_t)D * HC * 2);
  unsigned short* WePk    = (unsigned short*) carve((size_t)D * HC * 2);
  unsigned short* WlePk   = (unsigned short*) carve((size_t)HC * C * 2);
  unsigned short* relpbf  = (unsigned short*) carve((size_t)R * HC * 2);

  auto g1 = [](size_t n) { return dim3((unsigned)((n + 255) / 256)); };

  // ---- init (d_out/d_ws are poisoned by the harness; re-init every call) ----
  float negbig = -3.0e38f;
  unsigned bits; memcpy(&bits, &negbig, 4);
  unsigned initkey = bits ^ ((bits >> 31) ? 0xFFFFFFFFu : 0x80000000u);
  k_fill_f32<<<g1((size_t)N * 2), 256, 0, stream>>>(denom, 0.0f, (size_t)N * 2);
  k_fill_u32<<<g1((size_t)N * 2), 256, 0, stream>>>(amaxk, initkey, (size_t)N * 2);
  k_fill_f32<<<g1(R), 256, 0, stream>>>(counts, 0.0f, (size_t)R);
  k_fill_f32<<<g1((size_t)N * C), 256, 0, stream>>>(out_nodes, 0.0f, (size_t)N * C);

  // ---- bf16 conversions (A-side) and packed-fragment weights (B-side) ----
  k_cvt_bf16<<<g1((size_t)N * D), 256, 0, stream>>>(x,    xbf,   (size_t)N * D);
  k_cvt_bf16<<<g1((size_t)R * D), 256, 0, stream>>>(rels, relbf, (size_t)R * D);
  {
    int frag128 = (HC / 16) * (D  / 32) * 32;   // W_l / W_r / W_e : K=128, Nc=256
    int frag256 = (C  / 16) * (HC / 32) * 32;   // W_le            : K=256, Nc=128
    k_packB<<<g1(frag128), 256, 0, stream>>>(W_l,  WlPk,  D,  HC);
    k_packB<<<g1(frag128), 256, 0, stream>>>(W_r,  WrPk,  D,  HC);
    k_packB<<<g1(frag128), 256, 0, stream>>>(W_e,  WePk,  D,  HC);
    k_packB<<<g1(frag256), 256, 0, stream>>>(W_le, WlePk, HC, C);
  }

  // ---- dense GEMMs via v_wmma_f32_16x16x32_bf16 (K unrolled, 4 n-tiles/wave)
  {
    dim3 gN((unsigned)(((N + 15) / 16 + 7) / 8), (unsigned)(HC / 64));
    k_gemm_bf16<128, 4><<<gN, 256, 0, stream>>>(xbf, WlPk, b_l, xl, N, HC);
    k_gemm_bf16<128, 4><<<gN, 256, 0, stream>>>(xbf, WrPk, b_r, xr, N, HC);
    dim3 gR((unsigned)(((R + 15) / 16 + 7) / 8), (unsigned)(HC / 64));
    k_gemm_bf16<128, 4><<<gR, 256, 0, stream>>>(relbf, WePk, nullptr, relp, R, HC);
  }

  // ---- mean edge-attr projection: meanp = (1/E) * sum_r count[r]*relp[r] ----
  k_counts<<<g1(E), 256, 0, stream>>>(ridx, counts, E);
  k_meanp<<<1, 256, 0, stream>>>(counts, relp, meanp, R, 1.0f / (float)E);

  // ---- edge attention: logits, segment softmax, weighted scatter ----
  dim3 gwarp((unsigned)((EN + 7) / 8));   // 8 waves per 256-thread block
  k_alpha<<<gwarp, 256, 0, stream>>>(ei, ridx, xl, xr, relp, meanp, att, alpha, amaxk, E, EN);
  k_expsum<<<g1(EN), 256, 0, stream>>>(ei, alpha, amaxk, denom, E, EN);
  k_scatter<<<gwarp, 256, 0, stream>>>(ei, xl, alpha, denom, out_nodes, E, EN);
  k_biasadd<<<g1((size_t)N * C), 256, 0, stream>>>(out_nodes, bias_g, (size_t)N * C, C);

  // ---- relation branch: relu(relp) @ W_le + b_le ----
  k_relu_cvt_bf16<<<g1((size_t)R * HC), 256, 0, stream>>>(relp, relpbf, (size_t)R * HC);
  {
    dim3 gRel((unsigned)(((R + 15) / 16 + 7) / 8), (unsigned)(C / 64));
    k_gemm_bf16<256, 4><<<gRel, 256, 0, stream>>>(relpbf, WlePk, b_le, out_rel, R, C);
  }
}
